// MoE_Layer_79912161509414
// MI455X (gfx1250) — compile-verified, matrix operand order checked by reading, not run on hardware
//
#include <hip/hip_runtime.h>
#include <hip/hip_bf16.h>
#include <math.h>

// ---------------------------------------------------------------------------
// MoE layer for MI455X (gfx1250): sparse top-2 routing + bf16 WMMA expert MLPs
// with weights pre-packed into WMMA-native bf16 tiles in workspace.
// ---------------------------------------------------------------------------

#define TT 8192   // tokens = B*S
#define DD 1024   // model dim
#define EE 8      // experts
#define HH 4096   // expert hidden (4*D)
#define MT 16     // token rows per block tile
#define HC 128    // H chunk per iteration (8 waves x 16 cols)

typedef __attribute__((ext_vector_type(16))) __bf16          v16bf;
typedef __attribute__((ext_vector_type(16))) unsigned short  v16u;
typedef __attribute__((ext_vector_type(8)))  unsigned short  u16x8;
typedef __attribute__((ext_vector_type(8)))  float           v8f;

union BfPun { v16u u; v16bf b; };

static __device__ __forceinline__ unsigned short bfc(float f) {
    __bf16 h = (__bf16)f;                       // native v_cvt (RNE)
    return __builtin_bit_cast(unsigned short, h);
}

// Per-lane K mapping for 16-bit A/B operands (ISA 7.12.2):
//   j<8  -> k = half*8 + j          (K 0..15 block)
//   j>=8 -> k = 16 + half*8 + (j-8) (K 16..31 block)

// A operand from bf16 LDS row (two 16B contiguous segments).
static __device__ __forceinline__ v16bf load_a_lds(const unsigned short* __restrict__ row, int half) {
    u16x8 s0 = *(const u16x8*)(row + half * 8);
    u16x8 s1 = *(const u16x8*)(row + 16 + half * 8);
    BfPun r;
#pragma unroll
    for (int j = 0; j < 8; ++j) { r.u[j] = s0[j]; r.u[j + 8] = s1[j]; }
    return r.b;
}

// B operand from pre-packed tile: lane's 16 bf16 are contiguous (32B).
static __device__ __forceinline__ v16bf load_b_packed(const unsigned short* __restrict__ tile, int lane) {
    BfPun r;
    r.u = *(const v16u*)(tile + lane * 16);      // two global_load_b128
    return r.b;
}

// Fallback B operand straight from fp32 weights (native cvt casts).
static __device__ __forceinline__ v16bf load_b_f32(const float* __restrict__ p, int half) {
    float4 a = *(const float4*)(p + half * 8);
    float4 b = *(const float4*)(p + half * 8 + 4);
    float4 c = *(const float4*)(p + 16 + half * 8);
    float4 d = *(const float4*)(p + 16 + half * 8 + 4);
    BfPun r;
    r.b[0]  = (__bf16)a.x; r.b[1]  = (__bf16)a.y; r.b[2]  = (__bf16)a.z; r.b[3]  = (__bf16)a.w;
    r.b[4]  = (__bf16)b.x; r.b[5]  = (__bf16)b.y; r.b[6]  = (__bf16)b.z; r.b[7]  = (__bf16)b.w;
    r.b[8]  = (__bf16)c.x; r.b[9]  = (__bf16)c.y; r.b[10] = (__bf16)c.z; r.b[11] = (__bf16)c.w;
    r.b[12] = (__bf16)d.x; r.b[13] = (__bf16)d.y; r.b[14] = (__bf16)d.z; r.b[15] = (__bf16)d.w;
    return r.b;
}

// ---------------------------------------------------------------------------
__global__ void zero_kernel(float* __restrict__ out, int nOut, int* __restrict__ cnt) {
    int i = blockIdx.x * blockDim.x + threadIdx.x;
    if (i < nOut) out[i] = 0.0f;
    if (i < EE) cnt[i] = 0;
}

// ---------------------------------------------------------------------------
// Pack fp32 weight matrix [E*rows][cols] into per-16x32-tile, per-lane
// contiguous bf16 layout: Ws[tile*512 + slot*16 + j].
__global__ __launch_bounds__(256) void pack_weights(
    const float* __restrict__ W, unsigned short* __restrict__ Ws,
    int colTiles, int cols, long totalSlots)
{
    long gid = (long)blockIdx.x * 256 + threadIdx.x;   // slot id
    if (gid >= totalSlots) return;
    int slot = (int)(gid & 31);
    long tile = gid >> 5;
    int ct = (int)(tile % colTiles);
    long rowBlk = tile / colTiles;                     // global 16-row block
    int half = slot >> 4, n = slot & 15;
    const float* src = W + (rowBlk * 16 + n) * (long)cols + ct * 32;
    float4 a = *(const float4*)(src + half * 8);
    float4 b = *(const float4*)(src + half * 8 + 4);
    float4 c = *(const float4*)(src + 16 + half * 8);
    float4 d = *(const float4*)(src + 16 + half * 8 + 4);
    v16u o;
    o[0]  = bfc(a.x); o[1]  = bfc(a.y); o[2]  = bfc(a.z); o[3]  = bfc(a.w);
    o[4]  = bfc(b.x); o[5]  = bfc(b.y); o[6]  = bfc(b.z); o[7]  = bfc(b.w);
    o[8]  = bfc(c.x); o[9]  = bfc(c.y); o[10] = bfc(c.z); o[11] = bfc(c.w);
    o[12] = bfc(d.x); o[13] = bfc(d.y); o[14] = bfc(d.z); o[15] = bfc(d.w);
    *(v16u*)(Ws + tile * 512 + slot * 16) = o;
}

// ---------------------------------------------------------------------------
// Router: one wave32 per token. logits = x @ Wr^T + br, softmax, top-2,
// renormalize; scatter (token, weight) into per-expert lists.
__global__ __launch_bounds__(256) void router_kernel(
    const float* __restrict__ x, const float* __restrict__ Wr,
    const float* __restrict__ br, int* __restrict__ cnt,
    int* __restrict__ perm, float* __restrict__ wgt)
{
    __shared__ float sWr[EE * DD];  // 32 KB
    for (int i = threadIdx.x; i < EE * DD; i += 256) sWr[i] = Wr[i];
    __syncthreads();

    int wave = threadIdx.x >> 5;
    int lane = threadIdx.x & 31;
    int t = blockIdx.x * 8 + wave;
    if (t >= TT) return;

    const float* xr = x + (size_t)t * DD;
    float acc[EE];
#pragma unroll
    for (int e = 0; e < EE; ++e) acc[e] = 0.0f;
    for (int i = 0; i < DD / 32; ++i) {
        float xv = xr[i * 32 + lane];
#pragma unroll
        for (int e = 0; e < EE; ++e) acc[e] += xv * sWr[e * DD + i * 32 + lane];
    }
#pragma unroll
    for (int e = 0; e < EE; ++e) {
        float v = acc[e];
#pragma unroll
        for (int off = 16; off >= 1; off >>= 1) v += __shfl_xor(v, off, 32);
        acc[e] = v + br[e];
    }
    if (lane == 0) {
        float mx = acc[0];
        for (int e = 1; e < EE; ++e) mx = fmaxf(mx, acc[e]);
        float p[EE];
        for (int e = 0; e < EE; ++e) p[e] = expf(acc[e] - mx);
        int i1 = 0;
        for (int e = 1; e < EE; ++e) if (p[e] > p[i1]) i1 = e;
        int i2 = (i1 == 0) ? 1 : 0;
        for (int e = 0; e < EE; ++e) if (e != i1 && p[e] > p[i2]) i2 = e;
        float inv = 1.0f / (p[i1] + p[i2]);
        float w1 = p[i1] * inv, w2 = p[i2] * inv;
        int s1 = atomicAdd(&cnt[i1], 1);
        perm[i1 * TT + s1] = t; wgt[i1 * TT + s1] = w1;
        int s2 = atomicAdd(&cnt[i2], 1);
        perm[i2 * TT + s2] = t; wgt[i2 * TT + s2] = w2;
    }
}

// ---------------------------------------------------------------------------
// Fused expert MLP over gathered token tiles:
//   out[t] += w * ( gelu(x_t @ W1e^T + b1e) @ W2e^T + b2e )
template <bool PACKED>
__global__ __launch_bounds__(256) void expert_mlp_kernel(
    const float* __restrict__ x,
    const float* __restrict__ W1f, const float* __restrict__ b1,
    const float* __restrict__ W2f, const float* __restrict__ b2,
    const unsigned short* __restrict__ W1s, const unsigned short* __restrict__ W2s,
    const int* __restrict__ cnt, const int* __restrict__ perm,
    const float* __restrict__ wgt, float* __restrict__ out)
{
    int e = blockIdx.y;
    int n = cnt[e];
    int base = blockIdx.x * MT;
    if (base >= n) return;  // uniform early exit

    __shared__ unsigned short sX[MT][DD];  // 32 KB bf16 gathered tokens
    __shared__ unsigned short sH[MT][HC];  // 4 KB  bf16 hidden chunk
    __shared__ int   sTok[MT];
    __shared__ float sWgt[MT];

    int tid = threadIdx.x;
    if (tid < MT) {
        int r = base + tid;
        if (r < n) { sTok[tid] = perm[e * TT + r]; sWgt[tid] = wgt[e * TT + r]; }
        else       { sTok[tid] = -1;               sWgt[tid] = 0.0f; }
    }
    __syncthreads();

    // Gather X rows -> bf16 LDS (pad rows with zeros).
    for (int idx = tid * 4; idx < MT * DD; idx += 256 * 4) {
        int r = idx / DD, c = idx % DD;
        int tok = sTok[r];
        float4 v = make_float4(0.f, 0.f, 0.f, 0.f);
        if (tok >= 0) v = *(const float4*)(x + (size_t)tok * DD + c);
        sX[r][c + 0] = bfc(v.x); sX[r][c + 1] = bfc(v.y);
        sX[r][c + 2] = bfc(v.z); sX[r][c + 3] = bfc(v.w);
    }
    __syncthreads();

    int wv = tid >> 5;        // wave 0..7
    int lane = tid & 31;
    int half = lane >> 4;     // A/B layout half
    int l16 = lane & 15;

    v8f acc[8];               // wave's 8 output N-tiles (128 columns)
#pragma unroll
    for (int i = 0; i < 8; ++i)
#pragma unroll
        for (int j = 0; j < 8; ++j) acc[i][j] = 0.0f;

    for (int hb = 0; hb < HH / HC; ++hb) {
        int hbase = hb * HC;

        // ---- GEMM1: wave's 16x16 slice of h = X @ W1e^T, K = D = 1024
        const unsigned short* w1tile =
            W1s + ((size_t)(e * (HH / 16) + hb * 8 + wv) * (DD / 32)) * 512;
        const float* w1p =
            W1f + (size_t)e * HH * DD + (size_t)(hbase + wv * 16 + l16) * DD;

        if (PACKED && hb + 1 < HH / HC) {
            // stream next chunk's W1 tiles into L2 (global_prefetch_b8)
            const unsigned short* nxt =
                W1s + ((size_t)(e * (HH / 16) + (hb + 1) * 8 + wv) * (DD / 32)) * 512;
            __builtin_prefetch(nxt + (size_t)lane * 512, 0, 1);
        }

        v8f c1;
#pragma unroll
        for (int j = 0; j < 8; ++j) c1[j] = 0.0f;
#pragma unroll 4
        for (int dk = 0; dk < DD / 32; ++dk) {
            v16bf a = load_a_lds(&sX[l16][dk * 32], half);
            v16bf b = PACKED ? load_b_packed(w1tile + dk * 512, lane)
                             : load_b_f32(w1p + dk * 32, half);
            c1 = __builtin_amdgcn_wmma_f32_16x16x32_bf16(
                     false, a, false, b, (short)0, c1, false, false);
        }
        // bias + exact GELU, store bf16 hidden chunk
        float bias = b1[(size_t)e * HH + hbase + wv * 16 + l16];
#pragma unroll
        for (int v = 0; v < 8; ++v) {
            float hv = c1[v] + bias;
            hv = 0.5f * hv * (1.0f + erff(hv * 0.70710678118654752f));
            sH[v + 8 * half][wv * 16 + l16] = bfc(hv);
        }
        __syncthreads();

        // ---- GEMM2: acc += h_chunk @ W2e[:, hbase:hbase+128]^T
        v16bf A[4];
#pragma unroll
        for (int kc = 0; kc < 4; ++kc)
            A[kc] = load_a_lds(&sH[l16][kc * 32], half);

#pragma unroll
        for (int nt = 0; nt < 8; ++nt) {
            const unsigned short* w2tile =
                W2s + ((size_t)(e * (DD / 16) + wv * 8 + nt) * (HH / 32) + hb * 4) * 512;
            const float* w2p =
                W2f + (size_t)e * DD * HH + (size_t)(wv * 128 + nt * 16 + l16) * HH + hbase;
#pragma unroll
            for (int kc = 0; kc < 4; ++kc) {
                v16bf b = PACKED ? load_b_packed(w2tile + kc * 512, lane)
                                 : load_b_f32(w2p + kc * 32, half);
                acc[nt] = __builtin_amdgcn_wmma_f32_16x16x32_bf16(
                              false, A[kc], false, b, (short)0, acc[nt], false, false);
            }
        }
        __syncthreads();  // protect sH before next iteration overwrites it
    }

    // ---- Epilogue: out[t, d] += w * (acc + b2[d])  (exactly two commutative
    // adds per element onto a zeroed buffer -> bitwise deterministic)
#pragma unroll
    for (int nt = 0; nt < 8; ++nt) {
        int d = wv * 128 + nt * 16 + l16;
        float bv = b2[(size_t)e * DD + d];
#pragma unroll
        for (int v = 0; v < 8; ++v) {
            int m = v + 8 * half;
            if (base + m < n) {
                float contrib = sWgt[m] * (acc[nt][v] + bv);
                atomicAdd(out + (size_t)sTok[m] * DD + d, contrib);
            }
        }
    }
}

// ---------------------------------------------------------------------------
extern "C" void kernel_launch(void* const* d_in, const int* in_sizes, int n_in,
                              void* d_out, int out_size, void* d_ws, size_t ws_size,
                              hipStream_t stream) {
    (void)in_sizes; (void)n_in; (void)out_size;
    const float* x  = (const float*)d_in[0];
    const float* Wr = (const float*)d_in[1];
    const float* br = (const float*)d_in[2];
    const float* W1 = (const float*)d_in[3];
    const float* b1 = (const float*)d_in[4];
    const float* W2 = (const float*)d_in[5];
    const float* b2 = (const float*)d_in[6];
    float* out = (float*)d_out;

    char* ws = (char*)d_ws;
    int*   cnt  = (int*)ws;                                   // 8 ints (256B slot)
    int*   perm = (int*)(ws + 256);                           // E*T ints
    float* wgt  = (float*)(ws + 256 + (size_t)EE * TT * 4);   // E*T floats

    size_t listEnd = 256 + (size_t)EE * TT * 8;
    size_t w1off = (listEnd + 1023) & ~(size_t)1023;
    size_t w1bytes = (size_t)EE * HH * DD * 2;
    size_t w2off = w1off + w1bytes;
    size_t w2bytes = (size_t)EE * DD * HH * 2;
    size_t need = w2off + w2bytes;                            // ~134.7 MB

    unsigned short* W1s = (unsigned short*)(ws + w1off);
    unsigned short* W2s = (unsigned short*)(ws + w2off);
    bool packed = (ws_size >= need);

    zero_kernel<<<(TT * DD + 255) / 256, 256, 0, stream>>>(out, TT * DD, cnt);
    router_kernel<<<TT / 8, 256, 0, stream>>>(x, Wr, br, cnt, perm, wgt);

    dim3 grid((TT + MT - 1) / MT, EE);
    if (packed) {
        long slots1 = (long)EE * (HH / 16) * (DD / 32) * 32;
        pack_weights<<<(int)((slots1 + 255) / 256), 256, 0, stream>>>(W1, W1s, DD / 32, DD, slots1);
        long slots2 = (long)EE * (DD / 16) * (HH / 32) * 32;
        pack_weights<<<(int)((slots2 + 255) / 256), 256, 0, stream>>>(W2, W2s, HH / 32, HH, slots2);
        expert_mlp_kernel<true><<<grid, 256, 0, stream>>>(
            x, W1, b1, W2, b2, W1s, W2s, cnt, perm, wgt, out);
    } else {
        expert_mlp_kernel<false><<<grid, 256, 0, stream>>>(
            x, W1, b1, W2, b2, W1s, W2s, cnt, perm, wgt, out);
    }
}
